// PointMLPBackBone_78262894068230
// MI455X (gfx1250) — compile-verified, matrix operand order checked by reading, not run on hardware
//
#include <hip/hip_runtime.h>
#include <stdint.h>

// ---------------------------------------------------------------------------
// PointMLP backbone for MI455X (gfx1250, wave32, WMMA bf16 16x16x32)
// ---------------------------------------------------------------------------
#define Bb 8
#define Nn 4096
#define Ss 1024
#define Kk 24
#define Cc 64
#define OUTC 128
#define INC 131   // 2C+3

typedef __attribute__((ext_vector_type(16))) __bf16 v16bf;
typedef __attribute__((ext_vector_type(8)))  float  v8f;

__device__ inline v8f wmma_bf16(v16bf a, v16bf b, v8f c) {
  // D = A(16x32 bf16) x B(32x16 bf16) + C(16x16 f32)
  return __builtin_amdgcn_wmma_f32_16x16x32_bf16(false, a, false, b,
                                                 (short)0, c, false, false);
}

// Force a global (addrspace 1) 256-bit load for weight fragments so the
// backend emits global_load_b128 pairs instead of flat_load_b128.
__device__ inline v16bf load_global_v16(const __bf16* p) {
  const __attribute__((address_space(1))) v16bf* gp =
      (const __attribute__((address_space(1))) v16bf*)(uintptr_t)p;
  return *gp;
}

// ---------------------------------------------------------------------------
// Weight prep: fp32 [128 x Cin] -> bf16 A-fragment layout.
// Fragment element (mt, ks, lane, e): m = mt*16 + lane%16,
//   kb = 8*(lane/16), k = ks*32 + (e<8 ? kb+e : 16+kb+(e-8))   (ISA 7.12.2)
// ---------------------------------------------------------------------------
__global__ void swizzle_w(const float* __restrict__ W, int Cin, int KS,
                          __bf16* __restrict__ dst) {
  int idx = blockIdx.x * blockDim.x + threadIdx.x;
  int total = 8 * KS * 512;
  if (idx >= total) return;
  int e  = idx & 15;
  int L  = (idx >> 4) & 31;
  int rest = idx >> 9;
  int ks = rest % KS;
  int mt = rest / KS;
  int m  = mt * 16 + (L & 15);
  int kb = 8 * (L >> 4);
  int k  = ks * 32 + (e < 8 ? kb + e : 16 + kb + (e - 8));
  float v = (k < Cin) ? W[m * Cin + k] : 0.f;
  dst[idx] = (__bf16)v;
}

// Fold conv-bias + BN(eval) into scale/shift: y = acc*sc[m] + sh[m]
__global__ void bn_prep(const float* __restrict__ bias,
                        const float* __restrict__ bn, // [4,128] g,be,mu,var
                        float* __restrict__ sc, float* __restrict__ sh) {
  int m = threadIdx.x;
  float g   = bn[0 * 128 + m];
  float be  = bn[1 * 128 + m];
  float mu  = bn[2 * 128 + m];
  float var = bn[3 * 128 + m];
  float s = g * rsqrtf(var + 1e-5f);
  sc[m] = s;
  sh[m] = be + (bias[m] - mu) * s;
}

// ---------------------------------------------------------------------------
// Furthest point sampling: one block per batch, sequential scan in LDS.
// ---------------------------------------------------------------------------
__global__ void fps_kernel(const float* __restrict__ xyz, int* __restrict__ fps) {
  __shared__ float sdist[Nn];
  __shared__ float rv[256];
  __shared__ int   ri[256];
  __shared__ int   s_last;
  int b = blockIdx.x, t = threadIdx.x;
  const float* X = xyz + (size_t)b * Nn * 3;
  for (int i = t; i < Nn; i += 256) sdist[i] = 1e10f;
  if (t == 0) { fps[b * Ss + 0] = 0; s_last = 0; }
  __syncthreads();
  for (int s = 1; s < Ss; ++s) {
    int last = s_last;
    float px = X[last * 3 + 0], py = X[last * 3 + 1], pz = X[last * 3 + 2];
    float bm = -1.f; int bi = 0;
    for (int i = t; i < Nn; i += 256) {
      float dx = X[i * 3 + 0] - px;
      float dy = X[i * 3 + 1] - py;
      float dz = X[i * 3 + 2] - pz;
      float d  = dx * dx + dy * dy + dz * dz;
      float dd = fminf(sdist[i], d);
      sdist[i] = dd;
      if (dd > bm) { bm = dd; bi = i; }
    }
    rv[t] = bm; ri[t] = bi;
    __syncthreads();
    for (int off = 128; off > 0; off >>= 1) {
      if (t < off) {
        if (rv[t + off] > rv[t] ||
            (rv[t + off] == rv[t] && ri[t + off] < ri[t])) {
          rv[t] = rv[t + off]; ri[t] = ri[t + off];
        }
      }
      __syncthreads();
    }
    if (t == 0) { s_last = ri[0]; fps[b * Ss + s] = ri[0]; }
    __syncthreads();
  }
}

// ---------------------------------------------------------------------------
// kNN (K=24) + gather new_xyz/new_points. Thread per query, register top-K.
// ---------------------------------------------------------------------------
__global__ void knn_kernel(const float* __restrict__ xyz,
                           const float* __restrict__ points,
                           const int* __restrict__ fps,
                           int* __restrict__ knn,
                           float* __restrict__ new_xyz,
                           float* __restrict__ new_pts) {
  __shared__ float sx[Nn * 3];
  int b = blockIdx.y;
  int s = blockIdx.x * 128 + threadIdx.x;
  const float* X = xyz + (size_t)b * Nn * 3;
  for (int i = threadIdx.x; i < Nn * 3; i += 128) sx[i] = X[i];
  __syncthreads();
  int fi = fps[b * Ss + s];
  float qx = sx[fi * 3 + 0], qy = sx[fi * 3 + 1], qz = sx[fi * 3 + 2];
  float bd[Kk]; int bi[Kk];
#pragma unroll
  for (int j = 0; j < Kk; ++j) { bd[j] = 1e30f; bi[j] = 0; }
  for (int n = 0; n < Nn; ++n) {
    float dx = sx[n * 3 + 0] - qx;
    float dy = sx[n * 3 + 1] - qy;
    float dz = sx[n * 3 + 2] - qz;
    float d = dx * dx + dy * dy + dz * dz;
    if (d < bd[Kk - 1]) {
      bd[Kk - 1] = d; bi[Kk - 1] = n;
#pragma unroll
      for (int j = Kk - 1; j > 0; --j) {
        if (bd[j] < bd[j - 1]) {
          float tv = bd[j]; bd[j] = bd[j - 1]; bd[j - 1] = tv;
          int   ti = bi[j]; bi[j] = bi[j - 1]; bi[j - 1] = ti;
        }
      }
    }
  }
  size_t base = ((size_t)b * Ss + s) * Kk;
#pragma unroll
  for (int j = 0; j < Kk; ++j) knn[base + j] = bi[j];
  new_xyz[((size_t)b * Ss + s) * 3 + 0] = qx;
  new_xyz[((size_t)b * Ss + s) * 3 + 1] = qy;
  new_xyz[((size_t)b * Ss + s) * 3 + 2] = qz;
  const float* P  = points + ((size_t)b * Nn + fi) * Cc;
  float*       NP = new_pts + ((size_t)b * Ss + s) * Cc;
  for (int c = 0; c < Cc; ++c) NP[c] = P[c];
}

// ---------------------------------------------------------------------------
// Batch-wide std of diff (ddof=1): accumulate sum / sumsq per batch.
// ---------------------------------------------------------------------------
__global__ void std_partial(const float* __restrict__ xyz,
                            const float* __restrict__ points,
                            const int* __restrict__ knn,
                            const float* __restrict__ new_xyz,
                            const float* __restrict__ new_pts,
                            float* __restrict__ stats) {
  __shared__ float r1[256], r2[256];
  int t = threadIdx.x;
  int g = blockIdx.x * 256 + t;
  int b   = g / (Ss * Kk);
  int rem = g % (Ss * Kk);
  int s = rem / Kk, k = rem % Kk;
  int n = knn[((size_t)b * Ss + s) * Kk + k];
  const float* P  = points  + ((size_t)b * Nn + n) * Cc;
  const float* NP = new_pts + ((size_t)b * Ss + s) * Cc;
  float sum = 0.f, ss = 0.f;
  for (int c = 0; c < Cc; ++c) { float d = P[c] - NP[c]; sum += d; ss += d * d; }
  const float* X  = xyz     + ((size_t)b * Nn + n) * 3;
  const float* NX = new_xyz + ((size_t)b * Ss + s) * 3;
  for (int c = 0; c < 3; ++c) { float d = X[c] - NX[c]; sum += d; ss += d * d; }
  r1[t] = sum; r2[t] = ss;
  __syncthreads();
  for (int off = 128; off > 0; off >>= 1) {
    if (t < off) { r1[t] += r1[t + off]; r2[t] += r2[t + off]; }
    __syncthreads();
  }
  if (t == 0) {
    atomicAdd(&stats[2 * b + 0], r1[0]);
    atomicAdd(&stats[2 * b + 1], r2[0]);
  }
}

__global__ void std_final(const float* __restrict__ stats,
                          float* __restrict__ stdv) {
  int b = threadIdx.x;
  const float M = (float)(Ss * Kk * (Cc + 3));
  float sum = stats[2 * b], ss = stats[2 * b + 1];
  float var = (ss - sum * sum / M) / (M - 1.f);
  stdv[b] = sqrtf(fmaxf(var, 0.f));
}

// ---------------------------------------------------------------------------
// Epilogue (compile-time MODE): BN scale/shift + (relu | residual-relu) +
// pack next B-fragment in LDS. MODE: 0=transfer (set res), 1=conv1 (frag
// only), 2=conv2 (add residual, update res). NVALID==NS==32 folds guards.
// ---------------------------------------------------------------------------
template <int MODE, bool WFRG, int NS, int NVALID>
__device__ inline void epilogue(v8f acc, int mt, int nt, int lane,
                                const float* __restrict__ sc,
                                const float* __restrict__ sh,
                                float* __restrict__ res,
                                __bf16* __restrict__ nxt) {
  int nloc = lane & 15;
  int mh   = lane >> 4;
  int n    = nt * 16 + nloc;
#pragma unroll
  for (int r = 0; r < 8; ++r) {
    int m = mt * 16 + 8 * mh + r;
    float v = acc[r] * sc[m] + sh[m];
    float outv;
    if (MODE == 2) {
      float x = (n < NVALID) ? res[m * NS + n] : 0.f;
      outv = fmaxf(v + x, 0.f);
      if (n < NVALID) res[m * NS + n] = outv;
    } else {
      outv = fmaxf(v, 0.f);
      if (MODE == 0 && n < NVALID) res[m * NS + n] = outv;
    }
    if (WFRG) {
      float fv = (n < NVALID) ? outv : 0.f;
      int kstep = m >> 5, kkm = m & 31;
      int Lp = nloc + 16 * ((kkm >> 3) & 1);
      int e  = ((kkm >= 16) ? 8 : 0) + (kkm & 7);
      nxt[(((n >> 4) * 4 + kstep) * 32 + Lp) * 16 + e] = (__bf16)fv;
    }
  }
}

// One conv stage: 2 M-tiles x 2 N-tiles per wave, KS k-steps of bf16 WMMA.
template <int KS, int MODE, bool WFRG, int NS, int NVALID>
__device__ inline void run_stage(const __bf16* __restrict__ W,
                                 const __bf16* __restrict__ cur,
                                 __bf16* __restrict__ nxt,
                                 const float* __restrict__ sc,
                                 const float* __restrict__ sh,
                                 float* __restrict__ res,
                                 int lane, int wave) {
#pragma unroll
  for (int mt2 = 0; mt2 < 2; ++mt2) {
    int mt = wave * 2 + mt2;
    v8f acc0 = {};
    v8f acc1 = {};
#pragma unroll
    for (int ks = 0; ks < KS; ++ks) {
      v16bf a  = load_global_v16(W + ((mt * KS + ks) * 32 + lane) * 16);
      v16bf b0 = *(const v16bf*)(cur + ((0 * KS + ks) * 32 + lane) * 16);
      v16bf b1 = *(const v16bf*)(cur + ((1 * KS + ks) * 32 + lane) * 16);
      acc0 = wmma_bf16(a, b0, acc0);
      acc1 = wmma_bf16(a, b1, acc1);
    }
    epilogue<MODE, WFRG, NS, NVALID>(acc0, mt, 0, lane, sc, sh, res, nxt);
    epilogue<MODE, WFRG, NS, NVALID>(acc1, mt, 1, lane, sc, sh, res, nxt);
  }
}

// Prefetch next stage's weight fragments (32KB) across the block: one
// global_prefetch_b8 per thread, 256B stride.
__device__ inline void prefetch_weights(const __bf16* W, int tid) {
  __builtin_prefetch(W + (size_t)tid * 128, 0, 1);
}

// ---------------------------------------------------------------------------
// PreExtraction: one block per (b,s). Builds 131x24 feats in LDS, runs
// transfer + 2 res-blocks entirely via WMMA in LDS, maxpools over K=24.
// ---------------------------------------------------------------------------
__global__ void __launch_bounds__(128)
pre_kernel(const float* __restrict__ xyz, const float* __restrict__ points,
           const float* __restrict__ alpha, const float* __restrict__ beta,
           const int* __restrict__ knn, const float* __restrict__ new_pts,
           const float* __restrict__ new_xyz, const float* __restrict__ stdv,
           const __bf16* __restrict__ wf0, const __bf16* __restrict__ wf,
           const float* __restrict__ sc, const float* __restrict__ sh,
           float* __restrict__ x0) {
  __shared__ float s_feats[INC * Kk];
  __shared__ __align__(32) __bf16 s_frag[2][5 * 2 * 512];
  __shared__ float s_res[OUTC * Kk];
  __shared__ int   s_idx[Kk];
  __shared__ float s_mean[Cc + 3];

  int s = blockIdx.x, b = blockIdx.y;
  int tid = threadIdx.x, lane = tid & 31, wave = tid >> 5;
  size_t bs = (size_t)b * Ss + s;

  if (tid < Kk) s_idx[tid] = knn[bs * Kk + tid];
  if (tid >= 32 && tid < 32 + Cc) s_mean[tid - 32] = new_pts[bs * Cc + (tid - 32)];
  if (tid >= 96 && tid < 99)      s_mean[Cc + (tid - 96)] = new_xyz[bs * 3 + (tid - 96)];
  prefetch_weights(wf0, tid);
  __syncthreads();

  float inv_std = 1.f / (stdv[b] + 1e-5f);
  for (int i = tid; i < INC * Kk; i += 128) {
    int kk = i / Kk, n = i - kk * Kk;
    float val;
    if (kk < Cc + 3) {
      int pt = s_idx[n];
      float raw = (kk < Cc) ? points[((size_t)b * Nn + pt) * Cc + kk]
                            : xyz[((size_t)b * Nn + pt) * 3 + (kk - Cc)];
      float d = raw - s_mean[kk];
      val = alpha[kk] * (d * inv_std) + beta[kk];
    } else {
      val = s_mean[kk - (Cc + 3)]; // anchor feature (new_points)
    }
    s_feats[i] = val;
  }
  __syncthreads();

  // pack initial B-fragment buffer (KS=5, 131 padded to 160)
  for (int i = tid; i < 2 * 5 * 512; i += 128) {
    int e = i & 15, L = (i >> 4) & 31;
    int rest = i >> 9;
    int ks = rest % 5, nt = rest / 5;
    int kb = 8 * (L >> 4);
    int kk = ks * 32 + (e < 8 ? kb + e : 16 + kb + (e - 8));
    int n  = nt * 16 + (L & 15);
    float v = (kk < INC && n < Kk) ? s_feats[kk * Kk + n] : 0.f;
    s_frag[0][i] = (__bf16)v;
  }
  __syncthreads();

  // stage 0: transfer 131->128 (KS=5), then 2 residual blocks (KS=4)
  prefetch_weights(wf, tid);
  run_stage<5, 0, true, Kk, Kk>(wf0, s_frag[0], s_frag[1], sc, sh, s_res, lane, wave);
  __syncthreads();
  prefetch_weights(wf + 16384, tid);
  run_stage<4, 1, true, Kk, Kk>(wf, s_frag[1], s_frag[0], sc + 128, sh + 128,
                                s_res, lane, wave);
  __syncthreads();
  prefetch_weights(wf + 2 * 16384, tid);
  run_stage<4, 2, true, Kk, Kk>(wf + 16384, s_frag[0], s_frag[1], sc + 256,
                                sh + 256, s_res, lane, wave);
  __syncthreads();
  prefetch_weights(wf + 3 * 16384, tid);
  run_stage<4, 1, true, Kk, Kk>(wf + 2 * 16384, s_frag[1], s_frag[0], sc + 384,
                                sh + 384, s_res, lane, wave);
  __syncthreads();
  run_stage<4, 2, false, Kk, Kk>(wf + 3 * 16384, s_frag[0], s_frag[1], sc + 512,
                                 sh + 512, s_res, lane, wave);
  __syncthreads();

  // adaptive max pool over K
  if (tid < OUTC) {
    int m = tid;
    float mx = s_res[m * Kk];
#pragma unroll
    for (int n = 1; n < Kk; ++n) mx = fmaxf(mx, s_res[m * Kk + n]);
    x0[((size_t)b * OUTC + m) * Ss + s] = mx;
  }
}

// ---------------------------------------------------------------------------
// PosExtraction: block per (b, 32-column chunk); 2 res-blocks fused via WMMA.
// ---------------------------------------------------------------------------
__global__ void __launch_bounds__(128)
pos_kernel(const float* __restrict__ x0, const __bf16* __restrict__ wf,
           const float* __restrict__ sc, const float* __restrict__ sh,
           float* __restrict__ out) {
  __shared__ __align__(32) __bf16 s_frag[2][4 * 2 * 512];
  __shared__ float s_res[OUTC * 32];
  int b = blockIdx.y;
  int c0 = blockIdx.x * 32;
  int tid = threadIdx.x, lane = tid & 31, wave = tid >> 5;

  prefetch_weights(wf + 4 * 16384, tid);
  for (int i = tid; i < OUTC * 32; i += 128) {
    int m = i >> 5, nl = i & 31;
    s_res[i] = x0[((size_t)b * OUTC + m) * Ss + c0 + nl];
  }
  __syncthreads();
  for (int i = tid; i < 2 * 4 * 512; i += 128) {
    int e = i & 15, L = (i >> 4) & 31;
    int rest = i >> 9;
    int ks = rest & 3, nt = rest >> 2;
    int kb = 8 * (L >> 4);
    int kk = ks * 32 + (e < 8 ? kb + e : 16 + kb + (e - 8));
    int n  = nt * 16 + (L & 15);
    s_frag[0][i] = (__bf16)s_res[kk * 32 + n];
  }
  __syncthreads();

  // layers 5..8: two residual blocks
  prefetch_weights(wf + 5 * 16384, tid);
  run_stage<4, 1, true, 32, 32>(wf + 4 * 16384, s_frag[0], s_frag[1], sc + 5 * 128,
                                sh + 5 * 128, s_res, lane, wave);
  __syncthreads();
  prefetch_weights(wf + 6 * 16384, tid);
  run_stage<4, 2, true, 32, 32>(wf + 5 * 16384, s_frag[1], s_frag[0], sc + 6 * 128,
                                sh + 6 * 128, s_res, lane, wave);
  __syncthreads();
  prefetch_weights(wf + 7 * 16384, tid);
  run_stage<4, 1, true, 32, 32>(wf + 6 * 16384, s_frag[0], s_frag[1], sc + 7 * 128,
                                sh + 7 * 128, s_res, lane, wave);
  __syncthreads();
  run_stage<4, 2, false, 32, 32>(wf + 7 * 16384, s_frag[1], s_frag[0], sc + 8 * 128,
                                 sh + 8 * 128, s_res, lane, wave);
  __syncthreads();

  for (int i = tid; i < OUTC * 32; i += 128) {
    int m = i >> 5, nl = i & 31;
    out[((size_t)b * OUTC + m) * Ss + c0 + nl] = s_res[i];
  }
}

// ---------------------------------------------------------------------------
extern "C" void kernel_launch(void* const* d_in, const int* in_sizes, int n_in,
                              void* d_out, int out_size, void* d_ws,
                              size_t ws_size, hipStream_t stream) {
  (void)in_sizes; (void)n_in; (void)out_size; (void)ws_size;
  const float* xyz         = (const float*)d_in[0];
  const float* points      = (const float*)d_in[1];
  const float* alpha       = (const float*)d_in[2];
  const float* beta        = (const float*)d_in[3];
  const float* transfer_w  = (const float*)d_in[4];
  const float* transfer_b  = (const float*)d_in[5];
  const float* transfer_bn = (const float*)d_in[6];
  const float* pre_w1      = (const float*)d_in[7];
  const float* pre_b1      = (const float*)d_in[8];
  const float* pre_bn1     = (const float*)d_in[9];
  const float* pre_w2      = (const float*)d_in[10];
  const float* pre_b2      = (const float*)d_in[11];
  const float* pre_bn2     = (const float*)d_in[12];
  const float* pos_w1      = (const float*)d_in[13];
  const float* pos_b1      = (const float*)d_in[14];
  const float* pos_bn1     = (const float*)d_in[15];
  const float* pos_w2      = (const float*)d_in[16];
  const float* pos_b2      = (const float*)d_in[17];
  const float* pos_bn2     = (const float*)d_in[18];
  float* out = (float*)d_out;

  char* ws = (char*)d_ws;
  size_t off = 0;
  auto take = [&](size_t bytes) -> void* {
    void* p = ws + off;
    off = (off + bytes + 255) & ~(size_t)255;
    return p;
  };
  int*    fps_idx = (int*)   take((size_t)Bb * Ss * 4);
  float*  new_xyz = (float*) take((size_t)Bb * Ss * 3 * 4);
  float*  new_pts = (float*) take((size_t)Bb * Ss * Cc * 4);
  int*    knn_idx = (int*)   take((size_t)Bb * Ss * Kk * 4);
  float*  stats   = (float*) take(2 * Bb * 4);
  float*  stdv    = (float*) take(Bb * 4);
  float*  x0      = (float*) take((size_t)Bb * OUTC * Ss * 4);
  __bf16* wf0     = (__bf16*)take((size_t)8 * 5 * 512 * 2);
  __bf16* wf      = (__bf16*)take((size_t)8 * 16384 * 2);
  float*  sc      = (float*) take(9 * 128 * 4);
  float*  sh      = (float*) take(9 * 128 * 4);

  // ---- weight / BN prep ----
  swizzle_w<<<(8 * 5 * 512 + 255) / 256, 256, 0, stream>>>(transfer_w, INC, 5, wf0);
  const float* wsrc[8] = { pre_w1, pre_w2, pre_w1 + 16384, pre_w2 + 16384,
                           pos_w1, pos_w2, pos_w1 + 16384, pos_w2 + 16384 };
  for (int l = 0; l < 8; ++l)
    swizzle_w<<<(8 * 4 * 512 + 255) / 256, 256, 0, stream>>>(wsrc[l], 128, 4,
                                                             wf + (size_t)l * 16384);
  const float* bsrc[9]  = { transfer_b, pre_b1, pre_b2, pre_b1 + 128, pre_b2 + 128,
                            pos_b1, pos_b2, pos_b1 + 128, pos_b2 + 128 };
  const float* bnsrc[9] = { transfer_bn, pre_bn1, pre_bn2, pre_bn1 + 512, pre_bn2 + 512,
                            pos_bn1, pos_bn2, pos_bn1 + 512, pos_bn2 + 512 };
  for (int l = 0; l < 9; ++l)
    bn_prep<<<1, 128, 0, stream>>>(bsrc[l], bnsrc[l], sc + l * 128, sh + l * 128);

  // ---- grouping ----
  fps_kernel<<<Bb, 256, 0, stream>>>(xyz, fps_idx);
  knn_kernel<<<dim3(Ss / 128, Bb), 128, 0, stream>>>(xyz, points, fps_idx,
                                                     knn_idx, new_xyz, new_pts);
  hipMemsetAsync(stats, 0, 2 * Bb * 4, stream);
  std_partial<<<(Bb * Ss * Kk) / 256, 256, 0, stream>>>(xyz, points, knn_idx,
                                                        new_xyz, new_pts, stats);
  std_final<<<1, Bb, 0, stream>>>(stats, stdv);

  // ---- pre extraction (fused WMMA pipeline + maxpool) ----
  pre_kernel<<<dim3(Ss, Bb), 128, 0, stream>>>(xyz, points, alpha, beta, knn_idx,
                                               new_pts, new_xyz, stdv, wf0, wf,
                                               sc, sh, x0);
  // ---- pos extraction (fused WMMA res-blocks) ----
  pos_kernel<<<dim3(Ss / 32, Bb), 128, 0, stream>>>(x0, wf, sc, sh, out);
}
